// AttentionEncoder_49331994362649
// MI455X (gfx1250) — matrix operand order repaired and checked by reading, not compiled, not run on hardware
//
#include <hip/hip_runtime.h>
#include <hip/hip_bf16.h>

// ---------------------------------------------------------------------------
// Transformer encoder block on MI455X (gfx1250, wave32, WMMA bf16).
// Heavy math: v_wmma_f32_16x16x32_bf16. GEMM B-tiles double-buffered in LDS
// via GLOBAL_LOAD_ASYNC_TO_LDS_B128 (ASYNCcnt-tracked, partial waits).
// ---------------------------------------------------------------------------

#define N_B 4
#define S_LEN 1024
#define E_DIM 1024
#define H_HEADS 16
#define D_HEAD 64
#define FF_DIM 4096

typedef __attribute__((ext_vector_type(16))) __bf16 v16bf;
typedef __attribute__((ext_vector_type(8)))  __bf16 v8bf;
typedef __attribute__((ext_vector_type(8)))  float  v8f;

static __device__ __forceinline__ void async_copy_b128_to_lds(
    const void* gsrc, void* lds_dst)
{
    // VDST = per-lane LDS byte address (generic LDS pointer low 32 bits ==
    // workgroup-relative LDS offset per the CDNA5 aperture rules);
    // VADDR = 64-bit global address.
    unsigned           lds_off = (unsigned)(unsigned long long)lds_dst;
    unsigned long long gaddr   = (unsigned long long)gsrc;
    asm volatile("global_load_async_to_lds_b128 %0, %1, off"
                 :: "v"(lds_off), "v"(gaddr)
                 : "memory");
}

static __device__ __forceinline__ void wait_async_le0() {
#if __has_builtin(__builtin_amdgcn_s_wait_asynccnt)
    __builtin_amdgcn_s_wait_asynccnt(0);
#else
    asm volatile("s_wait_asynccnt 0" ::: "memory");
#endif
}

static __device__ __forceinline__ void wait_async_le2() {
#if __has_builtin(__builtin_amdgcn_s_wait_asynccnt)
    __builtin_amdgcn_s_wait_asynccnt(2);
#else
    asm volatile("s_wait_asynccnt 2" ::: "memory");
#endif
}

static __device__ __forceinline__ __bf16 f2bf(float f) {
    union { float f; unsigned u; } v; v.f = f;
    unsigned r = v.u + 0x7FFFu + ((v.u >> 16) & 1u);   // round-to-nearest-even
    unsigned short h = (unsigned short)(r >> 16);
    __bf16 out;
    __builtin_memcpy(&out, &h, 2);
    return out;
}

static __device__ __forceinline__ v8f wmma_bf16(v16bf a, v16bf b, v8f c) {
    return __builtin_amdgcn_wmma_f32_16x16x32_bf16(
        /*neg_a=*/false, a, /*neg_b=*/false, b,
        /*c_mod=*/(short)0, c, /*reuse_a=*/false, /*reuse_b=*/false);
}

// ---------------------------------------------------------------------------
// fp32 -> bf16 conversion (weights)
// ---------------------------------------------------------------------------
__global__ __launch_bounds__(256) void cvt_bf16_kernel(const float* __restrict__ in,
                                                       __bf16* __restrict__ out, int n) {
    int i = blockIdx.x * 256 + threadIdx.x;
    if (i < n) out[i] = f2bf(in[i]);
}

// ---------------------------------------------------------------------------
// Embedding: we = relu(x@w1+b1)@w2+b2 ; out[n,s,e] = we[n,s] + pos[s,e]
// One wave per row.
// ---------------------------------------------------------------------------
__global__ __launch_bounds__(256) void embed_kernel(
    const float* __restrict__ x,   const float* __restrict__ w1,
    const float* __restrict__ b1,  const float* __restrict__ w2,
    const float* __restrict__ b2,  const float* __restrict__ pos,
    float* __restrict__ outf, __bf16* __restrict__ outb)
{
    const int idx  = blockIdx.x * 8 + (threadIdx.x >> 5);   // row 0..4095
    const int s    = idx & (S_LEN - 1);
    const int lane = threadIdx.x & 31;

    float xr[6];
    #pragma unroll
    for (int i = 0; i < 6; ++i) xr[i] = x[idx * 6 + i];

    float acc = 0.f;
    for (int j = lane; j < 256; j += 32) {
        float h = b1[j];
        #pragma unroll
        for (int i = 0; i < 6; ++i) h += xr[i] * w1[i * 256 + j];
        h = fmaxf(h, 0.f);
        acc += h * w2[j];
    }
    #pragma unroll
    for (int m = 16; m >= 1; m >>= 1) acc += __shfl_xor(acc, m, 32);
    const float we = acc + b2[0];

    const size_t base = (size_t)idx * E_DIM;
    for (int e = lane; e < E_DIM; e += 32) {
        float v = we + pos[s * E_DIM + e];
        outf[base + e] = v;
        outb[base + e] = f2bf(v);
    }
}

// ---------------------------------------------------------------------------
// Flash attention, Q=K=V=qkv (bf16, (N,S,E), head layout h*64+d).
// Block = 4 waves, each wave owns a 16-row Q tile; K-tile = 64 keys so the
// per-row shuffle reductions are amortized over 2x the columns.
// grid = (N*H, S/64).
// ---------------------------------------------------------------------------
__global__ __launch_bounds__(128) void attn_kernel(const __bf16* __restrict__ qkv,
                                                   __bf16* __restrict__ ao)
{
    const int nh    = blockIdx.x;
    const int n     = nh / H_HEADS;
    const int h     = nh % H_HEADS;
    const int wave  = threadIdx.x >> 5;
    const int lane  = threadIdx.x & 31;
    const int r     = lane & 15;
    const int hi    = lane >> 4;
    const int qbase = blockIdx.y * 64 + wave * 16;
    const float scale = 0.03125f;                 // 1/sqrt(E) = 1/32

    // ---- Q tile as two A-fragments (16x32 each over head-dim) ----
    const __bf16* qrow = qkv + ((size_t)(n * S_LEN + qbase + r) * E_DIM + h * D_HEAD);
    v16bf aq0, aq1;
    {
        v8bf t0 = *(const v8bf*)(qrow +  0 + hi * 16);
        v8bf t1 = *(const v8bf*)(qrow +  8 + hi * 16);
        v8bf t2 = *(const v8bf*)(qrow + 32 + hi * 16);
        v8bf t3 = *(const v8bf*)(qrow + 40 + hi * 16);
        #pragma unroll
        for (int j = 0; j < 8; ++j) {
            aq0[j] = t0[j]; aq0[j + 8] = t1[j];
            aq1[j] = t2[j]; aq1[j + 8] = t3[j];
        }
    }

    float m[8], l[8];
    v8f o[4];
    #pragma unroll
    for (int i = 0; i < 8; ++i) { m[i] = -1e30f; l[i] = 0.f; }
    #pragma unroll
    for (int f = 0; f < 4; ++f)
        #pragma unroll
        for (int i = 0; i < 8; ++i) o[f][i] = 0.f;

    __shared__ __bf16 lds_p[4][16][66];           // 16 rows x 64 keys (+pad)

    for (int kt = 0; kt < S_LEN; kt += 64) {
        // ---- scores: S(16x64) = Q(16x64) @ K^T(64x64), 4 key groups ----
        v8f c[4];
        #pragma unroll
        for (int g = 0; g < 4; ++g) {
            const int key = kt + g * 16 + r;      // B-fragment col = lane&15
            const __bf16* krow = qkv + ((size_t)(n * S_LEN + key) * E_DIM + h * D_HEAD);
            v16bf b0 = *(const v16bf*)(krow +      hi * 16);   // d = 0..31
            v16bf b1 = *(const v16bf*)(krow + 32 + hi * 16);   // d = 32..63
            v8f cc;
            #pragma unroll
            for (int i = 0; i < 8; ++i) cc[i] = 0.f;
            cc = wmma_bf16(aq0, b0, cc);
            cc = wmma_bf16(aq1, b1, cc);
            c[g] = cc;
        }

        // ---- online softmax update (per-row stats via intra-half shuffles) ----
        #pragma unroll
        for (int i = 0; i < 8; ++i) {
            float s0 = c[0][i] * scale;
            float s1 = c[1][i] * scale;
            float s2 = c[2][i] * scale;
            float s3 = c[3][i] * scale;
            float tmax = fmaxf(fmaxf(s0, s1), fmaxf(s2, s3));
            #pragma unroll
            for (int mk = 8; mk >= 1; mk >>= 1)
                tmax = fmaxf(tmax, __shfl_xor(tmax, mk, 32));
            float mn = fmaxf(m[i], tmax);
            float al = __expf(m[i] - mn);
            float p0 = __expf(s0 - mn);
            float p1 = __expf(s1 - mn);
            float p2 = __expf(s2 - mn);
            float p3 = __expf(s3 - mn);
            float rs = (p0 + p1) + (p2 + p3);
            #pragma unroll
            for (int mk = 8; mk >= 1; mk >>= 1)
                rs += __shfl_xor(rs, mk, 32);
            l[i] = l[i] * al + rs;
            m[i] = mn;
            o[0][i] *= al; o[1][i] *= al; o[2][i] *= al; o[3][i] *= al;

            const int row = i + 8 * hi;           // C-layout row
            lds_p[wave][row][r]      = f2bf(p0);
            lds_p[wave][row][16 + r] = f2bf(p1);
            lds_p[wave][row][32 + r] = f2bf(p2);
            lds_p[wave][row][48 + r] = f2bf(p3);
        }

        asm volatile("s_wait_dscnt 0" ::: "memory");

        // ---- re-read P as two A-fragments (16 rows x 32 keys each) ----
        v16bf pa0, pa1;
        #pragma unroll
        for (int j = 0; j < 8; ++j) {
            pa0[j]     = lds_p[wave][r][hi * 16 + j];
            pa0[j + 8] = lds_p[wave][r][8 + hi * 16 + j];
            pa1[j]     = lds_p[wave][r][32 + hi * 16 + j];
            pa1[j + 8] = lds_p[wave][r][40 + hi * 16 + j];
        }

        // ---- O(16x64) += P(16x64) @ V(64x64) ----
        #pragma unroll
        for (int f = 0; f < 4; ++f) {
            v16bf bv0, bv1;
            #pragma unroll
            for (int j = 0; j < 16; ++j) {
                const int kk = kt + hi * 16 + j;  // contraction index = key
                bv0[j] = qkv[(size_t)(n * S_LEN + kk) * E_DIM + h * D_HEAD + f * 16 + r];
                bv1[j] = qkv[(size_t)(n * S_LEN + kk + 32) * E_DIM + h * D_HEAD + f * 16 + r];
            }
            o[f] = wmma_bf16(pa0, bv0, o[f]);
            o[f] = wmma_bf16(pa1, bv1, o[f]);
        }
    }

    // ---- normalize & store ao (bf16) ----
    #pragma unroll
    for (int f = 0; f < 4; ++f) {
        #pragma unroll
        for (int i = 0; i < 8; ++i) {
            float v = o[f][i] / l[i];
            const int row = qbase + i + 8 * hi;
            ao[(size_t)(n * S_LEN + row) * E_DIM + h * D_HEAD + f * 16 + r] = f2bf(v);
        }
    }
}

// ---------------------------------------------------------------------------
// bf16 GEMM: C = A(MxK) @ B(KxN) [+bias] [relu] [+res], stores f32 and/or bf16.
// Block 256 threads = 8 waves; tile 256x128 (each wave: 32 rows x 128 cols =
// 16 WMMA per K-step). B tiles double-buffered in LDS via async global->LDS
// B128 copies; s_wait_asynccnt 2 keeps the next tile's copies in flight
// while the current tile is consumed.
// ---------------------------------------------------------------------------
__global__ __launch_bounds__(256) void gemm_bf16_kernel(
    const __bf16* __restrict__ A, const __bf16* __restrict__ Bm,
    const float* __restrict__ bias, const float* __restrict__ res,
    float* __restrict__ outf, __bf16* __restrict__ outb,
    int Mdim, int Ndim, int Kdim, int relu)
{
    __shared__ __bf16 lds_b[2][32][136];   // ping-pong; rows 16B-aligned

    const int wave = threadIdx.x >> 5;
    const int lane = threadIdx.x & 31;
    const int r    = lane & 15;
    const int hi   = lane >> 4;
    const int n0   = blockIdx.x * 128;
    const int m0   = blockIdx.y * 256;
    const int mrow = m0 + wave * 32;
    const int tr   = threadIdx.x >> 4;          // 0..15
    const int tc   = (threadIdx.x & 15) * 8;    // 0..120
    const int nt   = Kdim >> 5;                 // #K-tiles (>= 2 always here)

    v8f acc[2][8];
    #pragma unroll
    for (int s = 0; s < 2; ++s)
        #pragma unroll
        for (int f = 0; f < 8; ++f)
            #pragma unroll
            for (int i = 0; i < 8; ++i) acc[s][f][i] = 0.f;

    // each thread stages 2 x 16B of the 32x128 B tile
    auto stage = [&](int tile, int buf) {
        #pragma unroll
        for (int rep = 0; rep < 2; ++rep) {
            const int kk = tr + rep * 16;
            const __bf16* src = Bm + (size_t)(tile * 32 + kk) * Ndim + n0 + tc;
            async_copy_b128_to_lds(src, &lds_b[buf][kk][tc]);
        }
    };

    // prologue: tiles 0 and 1 in flight
    stage(0, 0);
    stage(1, 1);

    for (int k = 0; k < nt; ++k) {
        const int buf = k & 1;
        // oldest 2 copies (this tile) complete; next tile may stay in flight
        if (k + 1 < nt) wait_async_le2(); else wait_async_le0();
        __syncthreads();

        // ---- two A fragments per wave (rows mrow..mrow+31) ----
        const int k0 = k * 32;
        v16bf a[2];
        #pragma unroll
        for (int s = 0; s < 2; ++s) {
            const __bf16* arow = A + (size_t)(mrow + s * 16 + r) * Kdim + k0;
            v8bf t0 = *(const v8bf*)(arow +      hi * 16);
            v8bf t1 = *(const v8bf*)(arow +  8 + hi * 16);
            #pragma unroll
            for (int j = 0; j < 8; ++j) { a[s][j] = t0[j]; a[s][j + 8] = t1[j]; }
        }

        #pragma unroll
        for (int f = 0; f < 8; ++f) {
            v16bf b;
            #pragma unroll
            for (int j = 0; j < 16; ++j) b[j] = lds_b[buf][hi * 16 + j][f * 16 + r];
            acc[0][f] = wmma_bf16(a[0], b, acc[0][f]);
            acc[1][f] = wmma_bf16(a[1], b, acc[1][f]);
        }
        __syncthreads();                 // all waves done reading lds_b[buf]

        if (k + 2 < nt) stage(k + 2, buf);
    }

    // ---- fused epilogue ----
    #pragma unroll
    for (int s = 0; s < 2; ++s) {
        #pragma unroll
        for (int f = 0; f < 8; ++f) {
            const int col = n0 + f * 16 + r;
            #pragma unroll
            for (int i = 0; i < 8; ++i) {
                const int row = mrow + s * 16 + i + 8 * hi;
                float v = acc[s][f][i];
                if (bias) v += bias[col];
                if (relu) v = fmaxf(v, 0.f);
                if (res)  v += res[(size_t)row * Ndim + col];
                if (outf) outf[(size_t)row * Ndim + col] = v;
                if (outb) outb[(size_t)row * Ndim + col] = f2bf(v);
            }
        }
    }
}

// ---------------------------------------------------------------------------
// LayerNorm over last dim (1024). One block per row.
// ---------------------------------------------------------------------------
__global__ __launch_bounds__(256) void ln_kernel(
    const float* __restrict__ in, const float* __restrict__ g,
    const float* __restrict__ b, float* __restrict__ outf,
    __bf16* __restrict__ outb)
{
    const size_t base = (size_t)blockIdx.x * E_DIM;
    const int tid  = threadIdx.x;
    const int lane = tid & 31;
    const int wave = tid >> 5;

    float s = 0.f, s2 = 0.f;
    for (int e = tid; e < E_DIM; e += 256) {
        float v = in[base + e];
        s += v; s2 += v * v;
    }
    #pragma unroll
    for (int mk = 16; mk >= 1; mk >>= 1) {
        s  += __shfl_xor(s, mk, 32);
        s2 += __shfl_xor(s2, mk, 32);
    }
    __shared__ float red[2][8];
    if (lane == 0) { red[0][wave] = s; red[1][wave] = s2; }
    __syncthreads();
    float ts = 0.f, ts2 = 0.f;
    #pragma unroll
    for (int w = 0; w < 8; ++w) { ts += red[0][w]; ts2 += red[1][w]; }
    const float mu   = ts * (1.0f / E_DIM);
    const float var  = ts2 * (1.0f / E_DIM) - mu * mu;
    const float rstd = rsqrtf(var + 1e-5f);

    for (int e = tid; e < E_DIM; e += 256) {
        float v = (in[base + e] - mu) * rstd * g[e] + b[e];
        outf[base + e] = v;
        if (outb) outb[base + e] = f2bf(v);
    }
}

// ---------------------------------------------------------------------------
// Final head: out[n] = tanh(dot(mp[n], out_w) + out_b)
// ---------------------------------------------------------------------------
__global__ __launch_bounds__(256) void head_kernel(
    const float* __restrict__ mp, const float* __restrict__ w,
    const float* __restrict__ b, float* __restrict__ out)
{
    const int n = blockIdx.x;
    const float* row = mp + (size_t)n * (S_LEN * E_DIM);
    float acc = 0.f;
    for (int i = threadIdx.x; i < S_LEN * E_DIM; i += 256)
        acc += row[i] * w[i];
    #pragma unroll
    for (int mk = 16; mk >= 1; mk >>= 1) acc += __shfl_xor(acc, mk, 32);
    __shared__ float red[8];
    if ((threadIdx.x & 31) == 0) red[threadIdx.x >> 5] = acc;
    __syncthreads();
    if (threadIdx.x == 0) {
        float t = 0.f;
        #pragma unroll
        for (int w8 = 0; w8 < 8; ++w8) t += red[w8];
        out[n] = tanhf(t + b[0]);
    }
}

// ---------------------------------------------------------------------------
// Host-side launcher
// ---------------------------------------------------------------------------
extern "C" void kernel_launch(void* const* d_in, const int* in_sizes, int n_in,
                              void* d_out, int out_size, void* d_ws, size_t ws_size,
                              hipStream_t stream)
{
    (void)in_sizes; (void)n_in; (void)out_size; (void)ws_size;

    const float* x       = (const float*)d_in[0];
    const float* we_w1   = (const float*)d_in[1];
    const float* we_b1   = (const float*)d_in[2];
    const float* we_w2   = (const float*)d_in[3];
    const float* we_b2   = (const float*)d_in[4];
    const float* pos_emb = (const float*)d_in[5];
    const float* attn_wo = (const float*)d_in[6];
    const float* attn_bo = (const float*)d_in[7];
    const float* ln1_g   = (const float*)d_in[8];
    const float* ln1_b   = (const float*)d_in[9];
    const float* ff_w1   = (const float*)d_in[10];
    const float* ff_b1   = (const float*)d_in[11];
    const float* ff_w2   = (const float*)d_in[12];
    const float* ff_b2   = (const float*)d_in[13];
    const float* ln2_g   = (const float*)d_in[14];
    const float* ln2_b   = (const float*)d_in[15];
    const float* out_w   = (const float*)d_in[16];
    const float* out_b   = (const float*)d_in[17];
    float* out = (float*)d_out;

    char* ws = (char*)d_ws;
    const size_t MB = 1024 * 1024;
    // liveness-checked aliasing plan (total 90 MB):
    float*  out_f32   = (float*) (ws +  0 * MB);   // A: 16MB
    __bf16* out_bf16  = (__bf16*)(ws + 16 * MB);   // B:  8MB
    __bf16* ao_bf16   = (__bf16*)(ws + 24 * MB);   // C:  8MB
    float*  wo_f32    = (float*) (ws + 32 * MB);   // D: 16MB
    float*  x1_f32    = (float*) (ws + 48 * MB);   // E: 16MB
    __bf16* x1_bf16   = (__bf16*)(ws + 64 * MB);   // F:  8MB
    __bf16* ff1_bf16  = (__bf16*)(ws + 16 * MB);   // G: 32MB (overlays B,C,D - dead)
    float*  ff2_f32   = (float*) (ws +  0 * MB);   // H: 16MB (overlays A - dead)
    float*  mp_f32    = (float*) (ws + 16 * MB);   // I: 16MB (overlays G head - dead)
    __bf16* wo_bf16   = (__bf16*)(ws + 72 * MB);   //  2MB
    __bf16* ffw1_bf16 = (__bf16*)(ws + 74 * MB);   //  8MB
    __bf16* ffw2_bf16 = (__bf16*)(ws + 82 * MB);   //  8MB

    // 0) weight conversion
    cvt_bf16_kernel<<<(E_DIM * E_DIM) / 256, 256, 0, stream>>>(attn_wo, wo_bf16, E_DIM * E_DIM);
    cvt_bf16_kernel<<<(E_DIM * FF_DIM) / 256, 256, 0, stream>>>(ff_w1, ffw1_bf16, E_DIM * FF_DIM);
    cvt_bf16_kernel<<<(FF_DIM * E_DIM) / 256, 256, 0, stream>>>(ff_w2, ffw2_bf16, FF_DIM * E_DIM);

    // 1) embedding + positional
    embed_kernel<<<(N_B * S_LEN) / 8, 256, 0, stream>>>(
        x, we_w1, we_b1, we_w2, we_b2, pos_emb, out_f32, out_bf16);

    // 2) flash attention
    attn_kernel<<<dim3(N_B * H_HEADS, S_LEN / 64), 128, 0, stream>>>(out_bf16, ao_bf16);

    // 3) attention output projection + bias + residual
    gemm_bf16_kernel<<<dim3(E_DIM / 128, (N_B * S_LEN) / 256), 256, 0, stream>>>(
        ao_bf16, wo_bf16, attn_bo, out_f32, wo_f32, (__bf16*)nullptr,
        N_B * S_LEN, E_DIM, E_DIM, 0);

    // 4) LayerNorm 1
    ln_kernel<<<N_B * S_LEN, 256, 0, stream>>>(wo_f32, ln1_g, ln1_b, x1_f32, x1_bf16);

    // 5) FF1 + bias + relu
    gemm_bf16_kernel<<<dim3(FF_DIM / 128, (N_B * S_LEN) / 256), 256, 0, stream>>>(
        x1_bf16, ffw1_bf16, ff_b1, (const float*)nullptr, (float*)nullptr, ff1_bf16,
        N_B * S_LEN, FF_DIM, E_DIM, 1);

    // 6) FF2 + bias + residual
    gemm_bf16_kernel<<<dim3(E_DIM / 128, (N_B * S_LEN) / 256), 256, 0, stream>>>(
        ff1_bf16, ffw2_bf16, ff_b2, x1_f32, ff2_f32, (__bf16*)nullptr,
        N_B * S_LEN, E_DIM, FF_DIM, 0);

    // 7) LayerNorm 2
    ln_kernel<<<N_B * S_LEN, 256, 0, stream>>>(ff2_f32, ln2_g, ln2_b, mp_f32, (__bf16*)nullptr);

    // 8) head: per-batch 1M dot + tanh
    head_kernel<<<N_B, 256, 0, stream>>>(mp_f32, out_w, out_b, out);
}
